// Recurrent_26439818674409
// MI455X (gfx1250) — matrix operand order repaired and checked by reading, not compiled, not run on hardware
//
#include <hip/hip_runtime.h>

// ---------------------------------------------------------------------------
// 2-layer LSTM, B=32, T=512, D=H=1024
// bf16 WMMA (v_wmma_f32_16x16x32_bf16) + async global->LDS staging of the
// shared A panels (x, h_prev), weights streamed from L2.
// ---------------------------------------------------------------------------

typedef __attribute__((ext_vector_type(16))) __bf16 v16bf;
typedef __attribute__((ext_vector_type(8)))  float  v8f;

namespace {
constexpr int kB = 32;
constexpr int kT = 512;
constexpr int kD = 1024;
constexpr int kH = 1024;
constexpr int kBH = kB * kH;              // 32768
constexpr int kWElems = 4 * kH * kD;      // 4096*1024 per weight matrix

constexpr int kRowG = 2048;               // global A row bytes (1024 bf16)
constexpr int kRowL = 2064;               // padded LDS row bytes (+16B: bank skew)
constexpr int kPanelL = 32 * kRowL;       // 66048 B per staged panel
constexpr int kGatesOff = 2 * kPanelL;    // fp32 gate tile offset in LDS
constexpr int kSmemBytes = kGatesOff + 4 * 32 * 16 * 4;  // 140288 B
}

union Frag16 {
    uint4 u[2];
    v16bf v;
};

// ---- CDNA5 async global->LDS copy (guarded), 16B per lane -----------------
#if defined(__has_builtin)
#if __has_builtin(__builtin_amdgcn_global_load_async_to_lds_b128)
#define LSTM_HAS_ASYNC_LDS 1
#endif
#if __has_builtin(__builtin_amdgcn_s_wait_asynccnt)
#define LSTM_HAS_WAIT_ASYNC 1
#endif
#endif

typedef int lstm_v4i __attribute__((vector_size(16)));
typedef __attribute__((address_space(1))) lstm_v4i as1_v4i;  // global
typedef __attribute__((address_space(3))) lstm_v4i as3_v4i;  // LDS

__device__ __forceinline__ void lstm_cp16(void* lds_dst, const void* gsrc) {
#if defined(LSTM_HAS_ASYNC_LDS)
    __builtin_amdgcn_global_load_async_to_lds_b128(
        (as1_v4i*)gsrc, (as3_v4i*)lds_dst, 0, 0);
#else
    *(uint4*)lds_dst = *(const uint4*)gsrc;
#endif
}

__device__ __forceinline__ void lstm_wait_async() {
#if defined(LSTM_HAS_ASYNC_LDS)
#if defined(LSTM_HAS_WAIT_ASYNC)
    __builtin_amdgcn_s_wait_asynccnt(0);
#else
    asm volatile("s_wait_asynccnt 0x0" ::: "memory");
#endif
#endif
}

// ---- fp32 -> bf16 flat conversion (weights) -------------------------------
__global__ void lstm_cvt_bf16(const float* __restrict__ src,
                              __bf16* __restrict__ dst, int n) {
    for (int i = blockIdx.x * blockDim.x + threadIdx.x; i < n;
         i += gridDim.x * blockDim.x)
        dst[i] = (__bf16)src[i];
}

// ---- inputs [B,T,D] fp32 -> [T,B,D] bf16 ----------------------------------
__global__ void lstm_cvt_x(const float* __restrict__ src,
                           __bf16* __restrict__ dst, int n) {
    for (int i = blockIdx.x * blockDim.x + threadIdx.x; i < n;
         i += gridDim.x * blockDim.x) {
        int d = i % kD;
        int t = (i / kD) % kT;
        int b = i / (kD * kT);
        dst[((size_t)t * kB + b) * kD + d] = (__bf16)src[i];
    }
}

// ---- state init: h0 -> bf16 ping buffer 0 + fp32 copy; c0 -> fp32 ---------
__global__ void lstm_init_state(const float* __restrict__ h0,
                                const float* __restrict__ c0,
                                __bf16* __restrict__ hb,   // [L][2][B*H]
                                float* __restrict__ hf,    // [L][B*H]
                                float* __restrict__ cf,    // [L][B*H]
                                int n)                      // n = L*B*H
{
    for (int i = blockIdx.x * blockDim.x + threadIdx.x; i < n;
         i += gridDim.x * blockDim.x) {
        int layer = i / kBH;
        int r = i % kBH;
        hb[(size_t)layer * 2 * kBH + r] = (__bf16)h0[i];  // buf 0 == parity t=0
        hf[i] = h0[i];
        cf[i] = c0[i];
    }
}

__global__ void lstm_copy_f32(const float* __restrict__ src,
                              float* __restrict__ dst, int n) {
    for (int i = blockIdx.x * blockDim.x + threadIdx.x; i < n;
         i += gridDim.x * blockDim.x)
        dst[i] = src[i];
}

// ---------------------------------------------------------------------------
// Fused LSTM step: gates = x*Wih^T + h*Whh^T + b, then cell update.
// Grid: 64 WGs (16-wide j slices). Block: 128 threads = 4 waves.
// Wave g computes gate g (i,f,g,o): C tile [32 x 16], K = 2048.
// x / h_prev panels staged once per WG into LDS (async), A-fragments from
// LDS (ds_load_b128), B-fragments (weight rows) streamed from global.
// ---------------------------------------------------------------------------
__global__ __launch_bounds__(128) void lstm_step_kernel(
    const __bf16* __restrict__ x,       // [32,1024] row-major bf16
    const __bf16* __restrict__ hprev,   // [32,1024] bf16
    __bf16* __restrict__ hnext,         // [32,1024] bf16
    const __bf16* __restrict__ Wih,     // [4096,1024] bf16 row-major
    const __bf16* __restrict__ Whh,     // [4096,1024] bf16 row-major
    const float* __restrict__ bias,     // [4096] fp32
    float* __restrict__ c,              // [32,1024] fp32, in-place
    float* __restrict__ hf,             // [32,1024] fp32 (for hN)
    float* __restrict__ out,            // [B,T,H] fp32 or nullptr
    int t)
{
    extern __shared__ char smem[];      // [panel x | panel h | gates f32]

    const int wave  = threadIdx.x >> 5;   // gate index 0..3 (i,f,g,o)
    const int lane  = threadIdx.x & 31;
    const int jbase = blockIdx.x * 16;

    // ---- stage A panels into LDS (16B chunks, padded rows) ----------------
    {
        const char* gx = (const char*)x;
        const char* gh = (const char*)hprev;
#pragma unroll 4
        for (int it = 0; it < 32; ++it) {
            const int chunk = it * 128 + threadIdx.x;   // 0..4095
            const int row = chunk >> 7;                 // 0..31
            const int col = (chunk & 127) << 4;         // byte offset in row
            lstm_cp16(smem + row * kRowL + col, gx + row * kRowG + col);
            lstm_cp16(smem + kPanelL + row * kRowL + col,
                      gh + row * kRowG + col);
        }
        lstm_wait_async();
        __syncthreads();
    }

    const int rowlo = lane & 15;          // A-tile row (M mod 16)
    const int khalf = lane >> 4;          // A-tile lane group
    const int ncol  = lane & 15;          // B-tile column (N)
    const int kgrp  = lane >> 4;          // B-tile K group

    v8f acc0 = {};   // M = 0..15
    v8f acc1 = {};   // M = 16..31

    const size_t wrow = (size_t)(wave * 1024 + jbase + ncol) * 1024;
    const __bf16* Wrow_ih = Wih + wrow;
    const __bf16* Wrow_hh = Whh + wrow;

    for (int phase = 0; phase < 2; ++phase) {
        const char* apanel = smem + (phase ? kPanelL : 0);
        const __bf16* Wr = phase ? Wrow_hh : Wrow_ih;
        const char* arow0 = apanel + (size_t)rowlo * kRowL;
        const char* arow1 = apanel + (size_t)(rowlo + 16) * kRowL;
#pragma unroll 4
        for (int kb = 0; kb < 1024; kb += 32) {
            Frag16 a0, a1, bf;
            // A layout (16-bit 16x32): e<8 -> K = khalf*8+e ; e>=8 -> 16+khalf*8+(e-8)
            a0.u[0] = *(const uint4*)(arow0 + (kb + khalf * 8) * 2);
            a0.u[1] = *(const uint4*)(arow0 + (kb + 16 + khalf * 8) * 2);
            a1.u[0] = *(const uint4*)(arow1 + (kb + khalf * 8) * 2);
            a1.u[1] = *(const uint4*)(arow1 + (kb + 16 + khalf * 8) * 2);
            // B layout (16-bit 32x16): lane = N, e<8 -> K = kgrp*8+e ; e>=8 -> 16+kgrp*8+(e-8)
            bf.u[0] = *(const uint4*)(Wr + kb + kgrp * 8);
            bf.u[1] = *(const uint4*)(Wr + kb + 16 + kgrp * 8);

            acc0 = __builtin_amdgcn_wmma_f32_16x16x32_bf16(
                false, a0.v, false, bf.v, (short)0, acc0, false, false);
            acc1 = __builtin_amdgcn_wmma_f32_16x16x32_bf16(
                false, a1.v, false, bf.v, (short)0, acc1, false, false);
        }
    }

    // C/D layout: VGPR r, lanes0-15 -> M=r, lanes16-31 -> M=r+8; N = lane&15
    float (*gsm)[32][16] = (float (*)[32][16])(smem + kGatesOff);
    const float bn = bias[wave * 1024 + jbase + ncol];
#pragma unroll
    for (int r = 0; r < 8; ++r) {
        const int m = r + khalf * 8;
        gsm[wave][m][ncol]      = acc0[r] + bn;
        gsm[wave][m + 16][ncol] = acc1[r] + bn;
    }
    __syncthreads();

    // Fused cell update: 512 elements over 128 threads
    for (int e = threadIdx.x; e < 32 * 16; e += 128) {
        const int b   = e >> 4;
        const int col = e & 15;
        const int j   = jbase + col;
        const int ci  = b * kH + j;

        const float iv = gsm[0][b][col];
        const float fv = gsm[1][b][col];
        const float gv = gsm[2][b][col];
        const float ov = gsm[3][b][col];

        const float si = 1.0f / (1.0f + __expf(-iv));
        const float sf = 1.0f / (1.0f + __expf(-fv));
        const float so = 1.0f / (1.0f + __expf(-ov));

        const float cn = sf * c[ci] + si * tanhf(gv);
        const float hn = so * tanhf(cn);

        c[ci]     = cn;
        hf[ci]    = hn;
        hnext[ci] = (__bf16)hn;
        if (out) out[((size_t)b * kT + t) * kH + j] = hn;
    }
}

// ---------------------------------------------------------------------------
extern "C" void kernel_launch(void* const* d_in, const int* in_sizes, int n_in,
                              void* d_out, int out_size, void* d_ws,
                              size_t ws_size, hipStream_t stream) {
    (void)in_sizes; (void)n_in; (void)out_size; (void)ws_size;

    const float* inputs = (const float*)d_in[0];
    const float* h0     = (const float*)d_in[1];
    const float* c0     = (const float*)d_in[2];
    const float* Wih0   = (const float*)d_in[3];
    const float* Whh0   = (const float*)d_in[4];
    const float* b0     = (const float*)d_in[5];
    const float* Wih1   = (const float*)d_in[6];
    const float* Whh1   = (const float*)d_in[7];
    const float* b1     = (const float*)d_in[8];
    float* out = (float*)d_out;

    // ---- workspace layout --------------------------------------------------
    char* ws = (char*)d_ws;
    __bf16* wb0 = (__bf16*)ws;                       // Wih0 bf16
    __bf16* wb1 = wb0 + kWElems;                     // Whh0
    __bf16* wb2 = wb1 + kWElems;                     // Wih1
    __bf16* wb3 = wb2 + kWElems;                     // Whh1
    __bf16* xb  = wb3 + kWElems;                     // [T][B][D] bf16
    __bf16* hb  = xb + (size_t)kT * kB * kD;         // [L][2][B*H] bf16
    float*  hf  = (float*)(hb + 2 * 2 * kBH);        // [L][B*H] fp32
    float*  cf  = hf + 2 * kBH;                      // [L][B*H] fp32

    // ---- one-time conversions ---------------------------------------------
    lstm_cvt_bf16<<<2048, 256, 0, stream>>>(Wih0, wb0, kWElems);
    lstm_cvt_bf16<<<2048, 256, 0, stream>>>(Whh0, wb1, kWElems);
    lstm_cvt_bf16<<<2048, 256, 0, stream>>>(Wih1, wb2, kWElems);
    lstm_cvt_bf16<<<2048, 256, 0, stream>>>(Whh1, wb3, kWElems);
    lstm_cvt_x<<<4096, 256, 0, stream>>>(inputs, xb, kB * kT * kD);
    lstm_init_state<<<256, 256, 0, stream>>>(h0, c0, hb, hf, cf, 2 * kBH);

    // ---- recurrence --------------------------------------------------------
    for (int t = 0; t < kT; ++t) {
        const int p  = t & 1;
        const int pn = (t + 1) & 1;
        const __bf16* x0  = xb + (size_t)t * kBH;    // B*D == kBH
        const __bf16* h0p = hb + 0 * 2 * kBH + (size_t)p  * kBH;
        __bf16*       h0n = hb + 0 * 2 * kBH + (size_t)pn * kBH;
        const __bf16* h1p = hb + 2 * kBH + (size_t)p  * kBH;
        __bf16*       h1n = hb + 2 * kBH + (size_t)pn * kBH;

        lstm_step_kernel<<<64, 128, kSmemBytes, stream>>>(
            x0, h0p, h0n, wb0, wb1, b0, cf, hf, nullptr, t);
        lstm_step_kernel<<<64, 128, kSmemBytes, stream>>>(
            h0n, h1p, h1n, wb2, wb3, b1, cf + kBH, hf + kBH, out, t);
    }

    // ---- hN, cN tails ------------------------------------------------------
    float* outHN = out + (size_t)kB * kT * kH;       // [L,B,H]
    float* outCN = outHN + 2 * kBH;                  // [L,B,H]
    lstm_copy_f32<<<256, 256, 0, stream>>>(hf, outHN, 2 * kBH);
    lstm_copy_f32<<<256, 256, 0, stream>>>(cf, outCN, 2 * kBH);
}
